// tae_77369540870647
// MI455X (gfx1250) — matrix operand order repaired and checked by reading, not compiled, not run on hardware
//
#include <hip/hip_runtime.h>

#define DEVINL __device__ __forceinline__

typedef __attribute__((ext_vector_type(16))) __bf16 v16bf;
typedef __attribute__((ext_vector_type(8)))  float  v8f;

union Frag {
    v16bf v;
    uint4 q[2];
    unsigned short us[16];
};

DEVINL unsigned short f2bf(float f) {
    unsigned int u = __float_as_uint(f);
    u += 0x7FFFu + ((u >> 16) & 1u);          // round-to-nearest-even
    return (unsigned short)(u >> 16);
}
DEVINL float bf2f(unsigned short h) {
    return __uint_as_float(((unsigned int)h) << 16);
}

// Fast gate nonlinearities: single TRANS ops where possible, branch-free.
DEVINL float fast_sigmoid(float x) {
    return __builtin_amdgcn_rcpf(1.0f + __expf(-x));
}
#if __has_builtin(__builtin_amdgcn_tanhf)
DEVINL float fast_tanh(float x) { return __builtin_amdgcn_tanhf(x); }
#elif __has_builtin(__builtin_amdgcn_tanh_f32)
DEVINL float fast_tanh(float x) { return __builtin_amdgcn_tanh_f32(x); }
#else
DEVINL float fast_tanh(float x) {
    // tanh(x) = 1 - 2/(exp(2x)+1); exp->inf / rcp->0 saturate to +/-1.
    float e = __expf(2.0f * x);
    return 1.0f - 2.0f * __builtin_amdgcn_rcpf(e + 1.0f);
}
#endif

// A-fragment 16x32 bf16 (ISA 7.12.2): per lane two 8-element K chunks at
// +khalf*8 and +16+khalf*8, row = lane&15.
DEVINL Frag load_frag_a(const unsigned short* p) {
    Frag f;
    f.q[0] = *(const uint4*)(p);
    f.q[1] = *(const uint4*)(p + 16);
    return f;
}
// B-fragment 32x16 bf16 from row-major [N][K]: per lane two 8-element K chunks at
// +khalf*8 and +16+khalf*8, row(column N) = lane&15.
DEVINL Frag load_frag_b(const unsigned short* p) {
    Frag f;
    f.q[0] = *(const uint4*)(p);
    f.q[1] = *(const uint4*)(p + 16);
    return f;
}
DEVINL v8f wmma_bf16(const Frag& a, const Frag& b, v8f c) {
    return __builtin_amdgcn_wmma_f32_16x16x32_bf16(
        false, a.v, false, b.v, (short)0, c, false, false);
}

// ---------------------------------------------------------------------------
// Prep kernel 1: fuse encoder linear into LSTM input weights.
// W_fused[g][i] = sum_c wih_e[g][c] * w_enc[c][i]   (bf16)
// bias_e[g]     = bih_e[g] + bhh_e[g] + sum_c wih_e[g][c] * b_enc[c]
// ---------------------------------------------------------------------------
__global__ void prep_fuse(const float* __restrict__ wih_e,
                          const float* __restrict__ w_enc,
                          const float* __restrict__ b_enc,
                          const float* __restrict__ bih_e,
                          const float* __restrict__ bhh_e,
                          unsigned short* __restrict__ wfuse,
                          float* __restrict__ bias_e) {
    int g = blockIdx.x * blockDim.x + threadIdx.x;
    if (g >= 1024) return;
    float bb = bih_e[g] + bhh_e[g];
    for (int c = 0; c < 256; ++c) bb += wih_e[g * 256 + c] * b_enc[c];
    bias_e[g] = bb;
    for (int i = 0; i < 32; ++i) {
        float s = 0.f;
        for (int c = 0; c < 256; ++c) s += wih_e[g * 256 + c] * w_enc[c * 32 + i];
        wfuse[g * 32 + i] = f2bf(s);
    }
}

// ---------------------------------------------------------------------------
// Prep kernel 2: bf16 weight conversions, decoder weight-sum and bias-sum.
// ---------------------------------------------------------------------------
__global__ void prep_conv(const float* __restrict__ whh_e,
                          const float* __restrict__ wih_d,
                          const float* __restrict__ whh_d,
                          const float* __restrict__ bih_d,
                          const float* __restrict__ bhh_d,
                          const float* __restrict__ w_dec,
                          unsigned short* __restrict__ whhe_bf,
                          unsigned short* __restrict__ whhd_bf,
                          unsigned short* __restrict__ wsum_bf,
                          unsigned short* __restrict__ wdec_bf,
                          float* __restrict__ bias_d) {
    int i = blockIdx.x * blockDim.x + threadIdx.x;
    int n = blockDim.x * gridDim.x;
    for (int idx = i; idx < 1024 * 256; idx += n) {
        whhe_bf[idx] = f2bf(whh_e[idx]);
        float wd = whh_d[idx];
        whhd_bf[idx] = f2bf(wd);
        wsum_bf[idx] = f2bf(wd + wih_d[idx]);
    }
    for (int idx = i; idx < 32 * 256; idx += n) wdec_bf[idx] = f2bf(w_dec[idx]);
    for (int idx = i; idx < 1024; idx += n) bias_d[idx] = bih_d[idx] + bhh_d[idx];
}

// ---------------------------------------------------------------------------
// Persistent LSTM autoencoder kernel: one WG (16 wave32s) owns 16 batch rows
// for all 512 recurrence steps. Wave w owns hidden columns [16w,16w+16) of
// every gate. h: bf16 in LDS; c: accumulator-layout VGPRs. Weight fragments
// are streamed from L2 each step; the K loop is kept as a real loop
// (#pragma unroll 1) so LICM cannot hoist 256+ VGPRs of weights and spill.
// ---------------------------------------------------------------------------
__global__ __launch_bounds__(512, 1) void lstm_ae_persistent(
    const float* __restrict__ x,             // [512][256][32] f32
    const unsigned short* __restrict__ wfuse,// [1024][32]  bf16
    const float* __restrict__ bias_e,        // [1024]
    const unsigned short* __restrict__ whhe, // [1024][256] bf16
    const unsigned short* __restrict__ whhd, // [1024][256] bf16
    const unsigned short* __restrict__ wsum, // [1024][256] bf16
    const unsigned short* __restrict__ wdec, // [32][256]   bf16
    const float* __restrict__ bias_d,        // [1024]
    const float* __restrict__ b_dec,         // [32]
    float* __restrict__ code_out,            // [512][256]
    float* __restrict__ out)                 // [512][256][32]
{
    constexpr int T = 256, INCH = 32, C = 256;
    __shared__ unsigned short h_lds[16][C];  // 8 KB: current h, bf16

    const int tid   = threadIdx.x;
    const int lane  = tid & 31;
    const int wave  = tid >> 5;                       // 0..15
    const int wfl   = __builtin_amdgcn_readfirstlane(wave);
    const int ln    = lane & 15;                      // tile column / frag row
    const int mb    = (lane >> 4) << 3;               // C/D row base: 0 or 8
    const int kofsA = (lane >> 4) << 3;               // A-frag k chunk base
    const int kofsB = (lane >> 4) << 3;               // B-frag k chunk base
    const int b0    = blockIdx.x << 4;                // 16 batch rows per WG
    const int ncol  = wave << 4;                      // hidden-column slice

    for (int i = tid; i < 16 * C; i += 512) (&h_lds[0][0])[i] = 0;
    v8f c_reg;
    #pragma unroll
    for (int e = 0; e < 8; ++e) c_reg[e] = 0.f;

    // per-lane base pointers (loop-invariant)
    const unsigned short* hA = &h_lds[0][0] + ln * C + kofsA;     // A frags of h
    const unsigned short* wE[4];
    const unsigned short* wF[4];
    const unsigned short* wD0[4];
    const unsigned short* wDs[4];
    float be[4], bd[4];
    #pragma unroll
    for (int g = 0; g < 4; ++g) {
        wE[g]  = whhe  + (size_t)(g * C + ncol + ln) * C    + kofsB;
        wDs[g] = wsum  + (size_t)(g * C + ncol + ln) * C    + kofsB;
        wD0[g] = whhd  + (size_t)(g * C + ncol + ln) * C    + kofsB;
        wF[g]  = wfuse + (size_t)(g * C + ncol + ln) * INCH + kofsB;
        be[g]  = bias_e[g * C + ncol + ln];
        bd[g]  = bias_d[g * C + ncol + ln];
    }
    const float* xbase = x + (size_t)(b0 + ln) * (T * INCH) + kofsA;
    __syncthreads();

    // ---------------- temporal encoder ----------------
    for (int t = 0; t < T; ++t) {
        // x_t tile: single K=32 A-fragment (f32 -> bf16 on the fly)
        Frag ax;
        {
            const float* xp = xbase + t * INCH;
            #pragma unroll
            for (int i = 0; i < 8; ++i) {
                ax.us[i]     = f2bf(xp[i]);
                ax.us[8 + i] = f2bf(xp[16 + i]);
            }
        }
        // prefetch next step's x slice (global_prefetch_b8, no counter cost)
        __builtin_prefetch(xbase + (t + 1) * INCH, 0, 3);

        v8f acc[4];
        #pragma unroll
        for (int g = 0; g < 4; ++g) {
            #pragma unroll
            for (int e = 0; e < 8; ++e) acc[g][e] = be[g];
            acc[g] = wmma_bf16(ax, load_frag_b(wF[g]), acc[g]);  // input term
        }
        // recurrent term: K = 256, kept as a real loop to bound live VGPRs
        #pragma unroll 1
        for (int kk = 0; kk < 8; ++kk) {
            Frag a = load_frag_a(hA + kk * 32);
            #pragma unroll
            for (int g = 0; g < 4; ++g)
                acc[g] = wmma_bf16(a, load_frag_b(wE[g] + kk * 32), acc[g]);
        }

        float hnew[8];
        #pragma unroll
        for (int e = 0; e < 8; ++e) {
            float ig = fast_sigmoid(acc[0][e]);
            float fg = fast_sigmoid(acc[1][e]);
            float gg = fast_tanh(acc[2][e]);
            float og = fast_sigmoid(acc[3][e]);
            float cn = fg * c_reg[e] + ig * gg;
            c_reg[e] = cn;
            hnew[e]  = og * fast_tanh(cn);
        }
        __syncthreads();                 // all waves done reading old h
        #pragma unroll
        for (int e = 0; e < 8; ++e)
            h_lds[mb + e][ncol + ln] = f2bf(hnew[e]);
        __syncthreads();                 // new h visible
    }

    // code_vec = h_T
    for (int i = tid; i < 16 * C; i += 512) {
        int r = i >> 8, cc = i & (C - 1);
        code_out[(size_t)(b0 + r) * C + cc] = bf2f(h_lds[r][cc]);
    }

    // ---------------- decoder ----------------
    // t==0: inp==0 -> gates = h @ whh_d.T + bias
    // t>=1: inp==h -> gates = h @ (wih_d + whh_d).T + bias   (precomputed wsum)
    const unsigned short* wP = wdec + (size_t)(((wave & 1) << 4) + ln) * C + kofsB;
    for (int t = 0; t < T; ++t) {
        v8f acc[4];
        #pragma unroll
        for (int g = 0; g < 4; ++g) {
            #pragma unroll
            for (int e = 0; e < 8; ++e) acc[g][e] = bd[g];
        }
        #pragma unroll 1
        for (int kk = 0; kk < 8; ++kk) {
            Frag a = load_frag_a(hA + kk * 32);
            #pragma unroll
            for (int g = 0; g < 4; ++g) {
                const unsigned short* wp = (t == 0) ? wD0[g] : wDs[g];
                acc[g] = wmma_bf16(a, load_frag_b(wp + kk * 32), acc[g]);
            }
        }

        float hnew[8];
        #pragma unroll
        for (int e = 0; e < 8; ++e) {
            float ig = fast_sigmoid(acc[0][e]);
            float fg = fast_sigmoid(acc[1][e]);
            float gg = fast_tanh(acc[2][e]);
            float og = fast_sigmoid(acc[3][e]);
            float cn = fg * c_reg[e] + ig * gg;
            c_reg[e] = cn;
            hnew[e]  = og * fast_tanh(cn);
        }
        __syncthreads();
        #pragma unroll
        for (int e = 0; e < 8; ++e)
            h_lds[mb + e][ncol + ln] = f2bf(hnew[e]);
        __syncthreads();

        // projection out_t = h @ w_dec.T + b_dec : 2 column tiles on waves 0/1.
        // readfirstlane keeps the branch scalar so EXEC stays all-ones.
        if (wfl < 2) {
            const float pb = b_dec[(wave << 4) + ln];
            v8f po;
            #pragma unroll
            for (int e = 0; e < 8; ++e) po[e] = pb;
            #pragma unroll 1
            for (int kk = 0; kk < 8; ++kk) {
                Frag a = load_frag_a(hA + kk * 32);
                po = wmma_bf16(a, load_frag_b(wP + kk * 32), po);
            }
            #pragma unroll
            for (int e = 0; e < 8; ++e)
                out[(size_t)(b0 + mb + e) * (T * INCH) + t * INCH + (wave << 4) + ln]
                    = po[e];
        }
    }
}

// ---------------------------------------------------------------------------
// Workspace layout (bytes, all 16B aligned):
//   0        wfuse  bf16 [1024][32]   (65536)
//   65536    bias_e f32  [1024]       (4096)
//   69632    whh_e  bf16 [1024][256]  (524288)
//   593920   whh_d  bf16 [1024][256]  (524288)
//   1118208  wsum   bf16 [1024][256]  (524288)
//   1642496  w_dec  bf16 [32][256]    (16384)
//   1658880  bias_d f32  [1024]       (4096)   total ~1.66 MB
// ---------------------------------------------------------------------------
extern "C" void kernel_launch(void* const* d_in, const int* in_sizes, int n_in,
                              void* d_out, int out_size, void* d_ws, size_t ws_size,
                              hipStream_t stream) {
    (void)in_sizes; (void)n_in; (void)out_size; (void)ws_size;
    const float* x     = (const float*)d_in[0];
    const float* w_enc = (const float*)d_in[1];
    const float* b_enc = (const float*)d_in[2];
    const float* wih_e = (const float*)d_in[3];
    const float* whh_e = (const float*)d_in[4];
    const float* bih_e = (const float*)d_in[5];
    const float* bhh_e = (const float*)d_in[6];
    const float* wih_d = (const float*)d_in[7];
    const float* whh_d = (const float*)d_in[8];
    const float* bih_d = (const float*)d_in[9];
    const float* bhh_d = (const float*)d_in[10];
    const float* w_dec = (const float*)d_in[11];
    const float* b_dec = (const float*)d_in[12];

    char* ws = (char*)d_ws;
    unsigned short* wfuse = (unsigned short*)(ws + 0);
    float*          biasE = (float*)(ws + 65536);
    unsigned short* whhe  = (unsigned short*)(ws + 69632);
    unsigned short* whhd  = (unsigned short*)(ws + 593920);
    unsigned short* wsumw = (unsigned short*)(ws + 1118208);
    unsigned short* wdecb = (unsigned short*)(ws + 1642496);
    float*          biasD = (float*)(ws + 1658880);

    float* code_out = (float*)d_out;
    float* out      = code_out + 512 * 256;

    prep_fuse<<<4, 256, 0, stream>>>(wih_e, w_enc, b_enc, bih_e, bhh_e, wfuse, biasE);
    prep_conv<<<128, 256, 0, stream>>>(whh_e, wih_d, whh_d, bih_d, bhh_d, w_dec,
                                       whhe, whhd, wsumw, wdecb, biasD);
    lstm_ae_persistent<<<32, 512, 0, stream>>>(x, wfuse, biasE, whhe, whhd, wsumw,
                                               wdecb, biasD, b_dec, code_out, out);
}
